// NAMHead_49984829391440
// MI455X (gfx1250) — matrix operand order repaired and checked by reading, not compiled, not run on hardware
//
#include <hip/hip_runtime.h>
#include <hip/hip_bf16.h>

#define B_  4096
#define F_  512
#define H1_ 64
#define H2_ 32
#define C_  100

typedef __attribute__((ext_vector_type(16))) _Float16 v16h;
typedef __attribute__((ext_vector_type(8)))  float    v8f;

// ---------------------------------------------------------------------------
// Kernel T: features [B,F] -> xT [F,B] so per-feature batch columns are
// contiguous (coalesced loads in the MLP kernel; avoids ~16x L2 amplification)
// ---------------------------------------------------------------------------
__global__ __launch_bounds__(256) void nam_transpose(const float* __restrict__ in,
                                                     float* __restrict__ out) {
    __shared__ float tile[32][33];
    const int tx = threadIdx.x & 31;
    const int ty = threadIdx.x >> 5;
    const int f0 = blockIdx.x * 32;   // along F (16 blocks)
    const int b0 = blockIdx.y * 32;   // along B (128 blocks)
#pragma unroll
    for (int i = 0; i < 32; i += 8)
        tile[ty + i][tx] = in[(b0 + ty + i) * F_ + f0 + tx];
    __syncthreads();
#pragma unroll
    for (int i = 0; i < 32; i += 8)
        out[(f0 + ty + i) * B_ + b0 + tx] = tile[tx][ty + i];
}

// ---------------------------------------------------------------------------
// Kernel 1: per-feature MLP (1->64->32->1), layer 2 on WMMA.
// Grid: 512 blocks (one feature each), 256 threads (8 waves), each wave
// handles 32 batch-tiles of 16 rows. All weight fragments are loop-invariant.
// WMMA uses 4 independent accumulators (inline C=0 -> no D->C RAW NOPs);
// b2 is added in the epilogue. Layer-3 N-reduction is a reduce-scatter
// butterfly: 8 bpermutes/tile instead of 32.
// ---------------------------------------------------------------------------
__global__ __launch_bounds__(256) void nam_feature_mlp(
    const float* __restrict__ xsrc,   // xT [F,B] if transposed, else features [B,F]
    int transposed,
    const float* __restrict__ w1,     // [F,64]
    const float* __restrict__ b1,     // [F,64]
    const float* __restrict__ w2,     // [F,64,32]
    const float* __restrict__ b2,     // [F,32]
    const float* __restrict__ w3,     // [F,32]
    const float* __restrict__ b3,     // [F]
    float* __restrict__ contrib)      // [B,F]
{
    const int f     = blockIdx.x;
    const int lane  = threadIdx.x & 31;
    const int wave  = threadIdx.x >> 5;
    const int khalf = lane >> 4;      // lane half selects K sub-block
    const int nlo   = lane & 15;

    // --- loop-invariant per-lane weights -----------------------------------
    // A-fragment K index pattern (16-bit A 16x32 layout):
    //   k(j) = (j&7) + ((j>>3)<<4) + khalf*8   (+32 for chunk 1)
    float w1v[32], b1v[32];
#pragma unroll
    for (int j = 0; j < 16; ++j) {
        const int k0 = (j & 7) + ((j >> 3) << 4) + khalf * 8;
        w1v[j]      = w1[f * H1_ + k0];
        w1v[16 + j] = w1[f * H1_ + k0 + 32];
        b1v[j]      = b1[f * H1_ + k0];
        b1v[16 + j] = b1[f * H1_ + k0 + 32];
    }
    // B fragments from w2: bf[ntile][kchunk]; lane holds column n = nlo+16*t,
    // K rows k = 32*c + (j&7) + ((j>>3)<<4) + 8*khalf (matches A-fragment K order).
    v16h bf[2][2];
#pragma unroll
    for (int t = 0; t < 2; ++t)
#pragma unroll
        for (int c = 0; c < 2; ++c) {
            const int n = nlo + 16 * t;
#pragma unroll
            for (int j = 0; j < 16; ++j) {
                const int k = 32 * c + (j & 7) + ((j >> 3) << 4) + khalf * 8;
                bf[t][c][j] = (_Float16)w2[(f * H1_ + k) * H2_ + n];
            }
        }
    const float c0init = b2[f * H2_ + nlo];        // added in epilogue
    const float c1init = b2[f * H2_ + nlo + 16];
    const float w3a = w3[f * H2_ + nlo];
    const float w3b = w3[f * H2_ + nlo + 16];
    const float b3v = b3[f];

    // reduce-scatter selectors / owned-row mapping (bit-reversal of nlo&7)
    const bool sel1 = (nlo & 1) != 0;
    const bool sel2 = (nlo & 2) != 0;
    const bool sel3 = (nlo & 4) != 0;
    const int  vown = ((nlo & 1) << 2) | (nlo & 2) | ((nlo >> 2) & 1);
    const int  rowown = vown + 8 * khalf;

    // --- batch-tile loop ----------------------------------------------------
#pragma unroll 1
    for (int bt = wave; bt < B_ / 16; bt += 8) {
        const int b0 = bt * 16;
        // x for this lane's A row (row m = nlo)
        const float x = transposed ? xsrc[f * B_ + b0 + nlo]
                                   : xsrc[(b0 + nlo) * F_ + f];
        if (transposed && bt + 8 < B_ / 16)
            __builtin_prefetch(&xsrc[f * B_ + b0 + 128 + nlo], 0, 3);

        // Layer 1 straight into A-fragment registers (f16, relu applied)
        v16h a0, a1;
#pragma unroll
        for (int j = 0; j < 16; ++j) {
            const float h0 = fmaxf(fmaf(x, w1v[j],      b1v[j]),      0.f);
            const float h1 = fmaxf(fmaf(x, w1v[16 + j], b1v[16 + j]), 0.f);
            a0[j] = (_Float16)h0;
            a1[j] = (_Float16)h1;
        }

        // Layer 2: 4 independent WMMAs, inline C=0 (no accumulation chains)
        v8f z = {};
        v8f d00 = __builtin_amdgcn_wmma_f32_16x16x32_f16(false, a0, false, bf[0][0],
                                                         (short)0, z, false, false);
        v8f d10 = __builtin_amdgcn_wmma_f32_16x16x32_f16(false, a0, false, bf[1][0],
                                                         (short)0, z, false, false);
        v8f d01 = __builtin_amdgcn_wmma_f32_16x16x32_f16(false, a1, false, bf[0][1],
                                                         (short)0, z, false, false);
        v8f d11 = __builtin_amdgcn_wmma_f32_16x16x32_f16(false, a1, false, bf[1][1],
                                                         (short)0, z, false, false);

        // Epilogue: h2 = relu(dA + dB + b2); t8 = h2_n*w3_n + h2_{n+16}*w3_{n+16}
        float t8[8];
#pragma unroll
        for (int v = 0; v < 8; ++v) {
            const float s0 = fmaxf((d00[v] + d01[v]) + c0init, 0.f);
            const float s1 = fmaxf((d10[v] + d11[v]) + c1init, 0.f);
            t8[v] = fmaf(s1, w3b, s0 * w3a);
        }

        // Reduce-scatter butterfly across the 16-lane N group: 8 bpermutes.
        float u[4];
#pragma unroll
        for (int i = 0; i < 4; ++i) {
            const float keep = sel1 ? t8[4 + i] : t8[i];
            const float send = sel1 ? t8[i]     : t8[4 + i];
            u[i] = keep + __shfl_xor(send, 1, 32);
        }
        float w[2];
#pragma unroll
        for (int i = 0; i < 2; ++i) {
            const float keep = sel2 ? u[2 + i] : u[i];
            const float send = sel2 ? u[i]     : u[2 + i];
            w[i] = keep + __shfl_xor(send, 2, 32);
        }
        {
            const float keep = sel3 ? w[1] : w[0];
            const float send = sel3 ? w[0] : w[1];
            float zr = keep + __shfl_xor(send, 4, 32);
            zr += __shfl_xor(zr, 8, 32);
            // lanes nlo<8 of each half each own one distinct row
            if (nlo < 8)
                contrib[(b0 + rowown) * F_ + f] = zr + b3v;
        }
    }
}

// ---------------------------------------------------------------------------
// Kernel 2: predictions = contributions @ w_out + b_out + bias  (WMMA, K=512)
// Grid: 256 M-tiles; 8 waves/block, waves 0..6 take N-tiles (N padded to 112)
// ---------------------------------------------------------------------------
__global__ __launch_bounds__(256) void nam_out_gemm(
    const float* __restrict__ contrib, // [B,512]
    const float* __restrict__ w_out,   // [512,100]
    const float* __restrict__ b_out,   // [100]
    const float* __restrict__ bias,    // [100]
    float* __restrict__ pred)          // [B,100]
{
    const int wave = threadIdx.x >> 5;
    if (wave >= 7) return;             // wave-uniform exit; EXEC stays full
    const int lane  = threadIdx.x & 31;
    const int khalf = lane >> 4;
    const int nlo   = lane & 15;
    const int n     = wave * 16 + nlo;
    const bool nvalid = (n < C_);

    const int b0 = blockIdx.x * 16;
    const float* arow = contrib + (b0 + nlo) * F_;

    v8f d;
    const float cinit = nvalid ? (b_out[n] + bias[n]) : 0.f;
#pragma unroll
    for (int v = 0; v < 8; ++v) d[v] = cinit;

#pragma unroll 2
    for (int c = 0; c < F_ / 32; ++c) {
        v16h a;
#pragma unroll
        for (int j = 0; j < 16; ++j) {
            const int k = 32 * c + (j & 7) + ((j >> 3) << 4) + khalf * 8;
            a[j] = (_Float16)arow[k];
        }
        v16h bfr;
#pragma unroll
        for (int j = 0; j < 16; ++j) {
            const int k = 32 * c + (j & 7) + ((j >> 3) << 4) + khalf * 8;
            bfr[j] = nvalid ? (_Float16)w_out[k * C_ + n] : (_Float16)0.f;
        }
        d = __builtin_amdgcn_wmma_f32_16x16x32_f16(false, a, false, bfr,
                                                   (short)0, d, false, false);
    }

    if (nvalid) {
#pragma unroll
        for (int v = 0; v < 8; ++v)
            pred[(b0 + v + 8 * khalf) * C_ + n] = d[v];
    }
}

// ---------------------------------------------------------------------------
extern "C" void kernel_launch(void* const* d_in, const int* in_sizes, int n_in,
                              void* d_out, int out_size, void* d_ws, size_t ws_size,
                              hipStream_t stream) {
    (void)in_sizes; (void)n_in; (void)out_size;
    const float* features = (const float*)d_in[0];
    const float* w1    = (const float*)d_in[1];
    const float* b1    = (const float*)d_in[2];
    const float* w2    = (const float*)d_in[3];
    const float* b2    = (const float*)d_in[4];
    const float* w3    = (const float*)d_in[5];
    const float* b3    = (const float*)d_in[6];
    const float* w_out = (const float*)d_in[7];
    const float* b_out = (const float*)d_in[8];
    const float* bias  = (const float*)d_in[9];

    float* pred    = (float*)d_out;            // [B, C]
    float* contrib = pred + (size_t)B_ * C_;   // [B, F]

    float* xT = (float*)d_ws;
    const int useT = (ws_size >= (size_t)B_ * F_ * sizeof(float)) ? 1 : 0;
    if (useT)
        nam_transpose<<<dim3(F_ / 32, B_ / 32), 256, 0, stream>>>(features, xT);

    nam_feature_mlp<<<F_, 256, 0, stream>>>(useT ? xT : features, useT,
                                            w1, b1, w2, b2, w3, b3, contrib);
    nam_out_gemm<<<B_ / 16, 256, 0, stream>>>(contrib, w_out, b_out, bias, pred);
}
